// StableNCA_13580686590086
// MI455X (gfx1250) — matrix operand order repaired and checked by reading, not compiled, not run on hardware
//
#include <hip/hip_runtime.h>

typedef __attribute__((ext_vector_type(16))) _Float16 v16h;
typedef __attribute__((ext_vector_type(8)))  _Float16 v8h;
typedef __attribute__((ext_vector_type(8)))  float    v8f;
typedef __attribute__((ext_vector_type(4)))  float    v4f;

#define NBAT 8
#define CH   28
#define HW   256
#define PIX  65536
#define NFRAG 78
// fragment region bases (1 KB per 32x16 f16 fragment)
#define FB_PW1 0
#define FB_PW2 32
#define FB_FW1 48
#define FB_SW1 64
#define FB_FW2 72
#define FB_SW2 76

__device__ __forceinline__ v8f wmma_f16(v16h a, v16h b, v8f c) {
  return __builtin_amdgcn_wmma_f32_16x16x32_f16(false, a, false, b, (short)0, c,
                                                false, false);
}

// A-fragment (16x32 f16) in ISA layout from a row-major [16 x 128] f16 LDS tile.
// lanes 0-15: M=lane, K = s*32 + {0..7, 16..23}; lanes 16-31: M=lane-16, K = +8.
__device__ __forceinline__ v16h afrag_lds(const _Float16* st, int lane, int s) {
  int m  = lane & 15;
  int kb = s * 32 + (lane >> 4) * 8;
  v8h lo = *(const v8h*)(st + m * 128 + kb);
  v8h hi = *(const v8h*)(st + m * 128 + kb + 16);
  return __builtin_shufflevector(lo, hi, 0, 1, 2, 3, 4, 5, 6, 7,
                                 8, 9, 10, 11, 12, 13, 14, 15);
}

// B-fragment: pre-packed so each lane reads its 32 contiguous bytes.
__device__ __forceinline__ v16h bfrag_g(const _Float16* wb, int frag, int lane) {
  return *(const v16h*)(wb + frag * 512 + lane * 16);
}

// ---------- weight packing into B-fragment layout ----------
__global__ void k_prep(const float* pw1, const float* pw2, const float* fW1,
                       const float* fW2, const float* sW1, const float* sW2,
                       _Float16* wfrag) {
  int t = blockIdx.x * blockDim.x + threadIdx.x;
  if (t >= NFRAG * 512) return;
  int frag = t >> 9;
  int e = t & 511;
  int k = e >> 4, n = e & 15;
  float v = 0.f;
  if (frag < FB_PW2) {            // pw1: K=128 (112 valid), N=128  [s][t]
    int f = frag - FB_PW1, s = f >> 3, tt = f & 7;
    int kg = s * 32 + k, ng = tt * 16 + n;
    if (kg < 112) v = pw1[kg * 128 + ng];
  } else if (frag < FB_FW1) {     // pw2: K=128, N=64
    int f = frag - FB_PW2, s = f >> 2, tt = f & 3;
    v = pw2[(s * 32 + k) * 64 + tt * 16 + n];
  } else if (frag < FB_SW1) {     // fW1 heads concatenated: K=64, N=4*32
    int f = frag - FB_FW1, s = f >> 3, tt = f & 7;
    int kg = s * 32 + k, j = tt * 16 + n;
    int a = j >> 5, h = j & 31;
    v = fW1[(a * 64 + kg) * 32 + h];
  } else if (frag < FB_FW2) {     // sW1 heads concatenated: K=64, N=4*16
    int f = frag - FB_SW1, s = f >> 2, tt = f & 3;
    int kg = s * 32 + k, j = tt * 16 + n;
    int a = j >> 4, h = j & 15;
    v = sW1[(a * 64 + kg) * 16 + h];
  } else if (frag < FB_SW2) {     // fW2 per head: K=32, N=16
    int a = frag - FB_FW2;
    v = fW2[(a * 32 + k) * 16 + n];
  } else {                        // sW2 pairs, block-diagonal 32x16
    int q = frag - FB_SW2;
    if (k < 16 && n < 8)        v = sW2[((2 * q) * 16 + k) * 8 + n];
    else if (k >= 16 && n >= 8) v = sW2[((2 * q + 1) * 16 + (k - 16)) * 8 + (n - 8)];
  }
  int lane = n + (((k >> 3) & 1) << 4);
  int j16  = (k & 7) + ((k >> 4) << 3);
  wfrag[frag * 512 + lane * 16 + j16] = (_Float16)v;
}

// ---------- alive-ratio reduction ----------
__global__ void k_init(float* hdr) {
  if (threadIdx.x < 8) hdr[threadIdx.x] = 0.f;
}

__global__ void k_count(const float* __restrict__ x, float* hdr) {
  int t = blockIdx.x * 256 + threadIdx.x;
  int b = t >> 16, p = t & 65535;
  int alive = (x[((size_t)b * CH + 3) * PIX + p] > 0.1f) ? 1 : 0;
  unsigned long long m = __ballot(alive);
  if ((threadIdx.x & 31) == 0) atomicAdd(&hdr[b], (float)__popcll(m));
}

__global__ void k_final(float* hdr) {
  int t = threadIdx.x;
  if (t < 8) hdr[8 + t] = hdr[t] * (1.f / 65536.f);
  if (t == 0) {
    float s = 0.f;
    for (int b = 0; b < 8; ++b) s += hdr[b];
    float ma = s / (8.f * 65536.f);
    int sel; float fac;
    if (ma < 0.15f)      { sel = 0; fac = 2.0f; }
    else if (ma > 0.6f)  { sel = 1; fac = 0.5f; }
    else                 { sel = 2; fac = 1.0f; }
    hdr[16] = (float)sel;
    hdr[17] = fac;
  }
}

// ---------- fused NCA step: conv + 6 WMMA GEMM stages + gating ----------
__global__ __launch_bounds__(256) void k_main(
    const float* __restrict__ x, const float* pb1, const float* pb2,
    const float* faw, const float* fab, const float* saw, const float* sab,
    const float* atemp, const float* fb1, const float* fb2,
    const float* sb1, const float* sb2,
    const float* stw1, const float* stb1, const float* stw2, const float* stb2,
    const float* stw3, const float* stb3,
    const _Float16* __restrict__ wfrag, const float* __restrict__ hdr,
    float* __restrict__ out) {
  __shared__ __align__(16) _Float16 s_stage[8 * 2048];  // 16x128 f16 per wave
  __shared__ __align__(16) float    s_fw[8 * 64];
  __shared__ __align__(16) float    s_sw[8 * 64];
  __shared__ __align__(16) float    s_ob[8 * 24 * 16];  // transposed updates

  const int lane = threadIdx.x & 31;
  const int w    = threadIdx.x >> 5;
  _Float16* st = s_stage + w * 2048;
  float* fwb = s_fw + w * 64;
  float* swb = s_sw + w * 64;
  float* ob  = s_ob + w * 384;

  const int gt = blockIdx.x * 8 + w;       // 32768 tiles of 16 pixels
  const int b  = gt >> 12;
  const int p0 = (gt & 4095) << 4;
  const float* xb = x + (size_t)b * CH * PIX;
  const int n  = lane & 15;
  const int mb = (lane >> 4) * 8;

  // --- depthwise 3x3 conv (ident/sobx/soby/lapl) into stage[16][128] f16 ---
  for (int q = 0; q < 14; ++q) {
    int id = lane + 32 * q;                // 448 (channel,pixel) pairs
    int c = id >> 4, px = id & 15;
    int p = p0 + px, yy = p >> 8, xx = p & 255;
    const float* xc = xb + c * PIX;
    float vv[9];
#pragma unroll
    for (int dy = 0; dy < 3; ++dy)
#pragma unroll
      for (int dx = 0; dx < 3; ++dx) {
        int y2 = yy + dy - 1, x2 = xx + dx - 1;
        vv[dy * 3 + dx] = (y2 >= 0 && y2 < HW && x2 >= 0 && x2 < HW)
                              ? xc[y2 * HW + x2] : 0.f;
      }
    float ident = vv[4];
    float sobx = (vv[2] + 2.f * vv[5] + vv[8]) - (vv[0] + 2.f * vv[3] + vv[6]);
    float soby = (vv[6] + 2.f * vv[7] + vv[8]) - (vv[0] + 2.f * vv[1] + vv[2]);
    float sum9 = vv[0] + vv[1] + vv[2] + vv[3] + vv[4] + vv[5] + vv[6] + vv[7] + vv[8];
    float lapl = sum9 - 9.f * vv[4];
    _Float16* r = st + px * 128;
    r[c]      = (_Float16)ident;
    r[28 + c] = (_Float16)sobx;
    r[56 + c] = (_Float16)soby;
    r[84 + c] = (_Float16)lapl;
  }
  {  // zero-pad K = 112..127
    v8h z = {};
    *(v8h*)(st + n * 128 + 112 + mb) = z;
  }

  // --- GEMM1: perc(16x128) @ pw1 -> relu -> stage ---
  v16h a1[4];
#pragma unroll
  for (int s = 0; s < 4; ++s) a1[s] = afrag_lds(st, lane, s);
#pragma unroll
  for (int t = 0; t < 8; ++t) {
    v8f c = {};
#pragma unroll
    for (int s = 0; s < 4; ++s)
      c = wmma_f16(a1[s], bfrag_g(wfrag, FB_PW1 + s * 8 + t, lane), c);
    float bias = pb1[t * 16 + n];
#pragma unroll
    for (int i = 0; i < 8; ++i)
      st[(mb + i) * 128 + t * 16 + n] = (_Float16)fmaxf(c[i] + bias, 0.f);
  }

  // --- GEMM2: @ pw2 -> proc(16x64); keep proc A-frags in registers ---
  v16h a2[4];
#pragma unroll
  for (int s = 0; s < 4; ++s) a2[s] = afrag_lds(st, lane, s);
#pragma unroll
  for (int t = 0; t < 4; ++t) {
    v8f c = {};
#pragma unroll
    for (int s = 0; s < 4; ++s)
      c = wmma_f16(a2[s], bfrag_g(wfrag, FB_PW2 + s * 4 + t, lane), c);
    float bias = pb2[t * 16 + n];
#pragma unroll
    for (int i = 0; i < 8; ++i)
      st[(mb + i) * 128 + t * 16 + n] = (_Float16)(c[i] + bias);
  }
  v16h ap[2];
  ap[0] = afrag_lds(st, lane, 0);
  ap[1] = afrag_lds(st, lane, 1);

  // --- per-pixel scalars (lanes 0..15): softmax gates, alive_ext, scale ---
  const float invT = 1.0f / atemp[0];
  float aext = 0.f, scl = 1.f;
  if (lane < 16) {
    int p = p0 + lane, yy = p >> 8, xx = p & 255;
    float hv[4], lg[4];
#pragma unroll
    for (int j = 0; j < 4; ++j) hv[j] = xb[(24 + j) * PIX + p];
#pragma unroll
    for (int a = 0; a < 4; ++a) {
      float s_ = fab[a];
#pragma unroll
      for (int j = 0; j < 4; ++j) s_ += hv[j] * faw[j * 4 + a];
      lg[a] = s_ * invT;
    }
    float mx = fmaxf(fmaxf(lg[0], lg[1]), fmaxf(lg[2], lg[3]));
    float e0 = __expf(lg[0] - mx), e1 = __expf(lg[1] - mx);
    float e2 = __expf(lg[2] - mx), e3 = __expf(lg[3] - mx);
    float inv = 1.f / (e0 + e1 + e2 + e3);
    v4f fv = {e0 * inv, e1 * inv, e2 * inv, e3 * inv};
    *(v4f*)(fwb + lane * 4) = fv;

    float mv[8];
#pragma unroll
    for (int j = 0; j < 8; ++j) mv[j] = xb[(16 + j) * PIX + p];
#pragma unroll
    for (int a = 0; a < 4; ++a) {
      float s_ = sab[a];
#pragma unroll
      for (int j = 0; j < 8; ++j) s_ += mv[j] * saw[j * 4 + a];
      lg[a] = s_ * invT;
    }
    mx = fmaxf(fmaxf(lg[0], lg[1]), fmaxf(lg[2], lg[3]));
    e0 = __expf(lg[0] - mx); e1 = __expf(lg[1] - mx);
    e2 = __expf(lg[2] - mx); e3 = __expf(lg[3] - mx);
    inv = 1.f / (e0 + e1 + e2 + e3);
    v4f sv = {e0 * inv, e1 * inv, e2 * inv, e3 * inv};
    *(v4f*)(swb + lane * 4) = sv;

    float mem_mean = (mv[0] + mv[1] + mv[2] + mv[3] +
                      mv[4] + mv[5] + mv[6] + mv[7]) * 0.125f;

    const float* x3 = xb + 3 * PIX;   // alive_ext = maxpool3(mask)
#pragma unroll
    for (int dy = 0; dy < 3; ++dy)
#pragma unroll
      for (int dx = 0; dx < 3; ++dx) {
        int y2 = yy + dy - 1, x2 = xx + dx - 1;
        if (y2 >= 0 && y2 < HW && x2 >= 0 && x2 < HW)
          aext = fmaxf(aext, (x3[y2 * HW + x2] > 0.1f) ? 1.f : 0.f);
      }

    float arb = hdr[8 + b];
    int sel   = (int)hdr[16];
    float fac = hdr[17];
    float h1[32];
#pragma unroll
    for (int kk = 0; kk < 32; ++kk)
      h1[kk] = fmaxf(mem_mean * stw1[kk] + arb * stw1[32 + kk] + stb1[kk], 0.f);
    float o = stb3[sel];
#pragma unroll
    for (int jj = 0; jj < 16; ++jj) {
      float s_ = stb2[jj];
#pragma unroll
      for (int kk = 0; kk < 32; ++kk) s_ += h1[kk] * stw2[kk * 16 + jj];
      o += fmaxf(s_, 0.f) * stw3[jj * 3 + sel];
    }
    scl = fac / (1.f + __expf(-o));
  }

  // --- GEMM3: proc @ fW1cat -> fh(16x128) relu -> stage ---
#pragma unroll
  for (int t = 0; t < 8; ++t) {
    v8f c = {};
#pragma unroll
    for (int s = 0; s < 2; ++s)
      c = wmma_f16(ap[s], bfrag_g(wfrag, FB_FW1 + s * 8 + t, lane), c);
    float bias = fb1[t * 16 + n];
#pragma unroll
    for (int i = 0; i < 8; ++i)
      st[(mb + i) * 128 + t * 16 + n] = (_Float16)fmaxf(c[i] + bias, 0.f);
  }

  // --- fast heads: fast_update = sum_a (fh_a @ fW2_a + fb2_a) * fast_w_a ---
  v4f fw4[8];
#pragma unroll
  for (int i = 0; i < 8; ++i) fw4[i] = *(const v4f*)(fwb + (mb + i) * 4);
  v8f afu = {};
#pragma unroll
  for (int a = 0; a < 4; ++a) {
    v16h af = afrag_lds(st, lane, a);    // head a = k-step a
    v8f c = {};
    c = wmma_f16(af, bfrag_g(wfrag, FB_FW2 + a, lane), c);
    float bias = fb2[a * 16 + n];
#pragma unroll
    for (int i = 0; i < 8; ++i) afu[i] += (c[i] + bias) * fw4[i][a];
  }

  // --- GEMM5: proc @ sW1cat -> sh(16x64) relu -> stage cols 0..63 ---
#pragma unroll
  for (int t = 0; t < 4; ++t) {
    v8f c = {};
#pragma unroll
    for (int s = 0; s < 2; ++s)
      c = wmma_f16(ap[s], bfrag_g(wfrag, FB_SW1 + s * 4 + t, lane), c);
    float bias = sb1[t * 16 + n];
#pragma unroll
    for (int i = 0; i < 8; ++i)
      st[(mb + i) * 128 + t * 16 + n] = (_Float16)fmaxf(c[i] + bias, 0.f);
  }

  // --- slow heads: two block-diagonal WMMAs cover 4 heads; xor-8 reduce ---
  v4f sw4[8];
#pragma unroll
  for (int i = 0; i < 8; ++i) sw4[i] = *(const v4f*)(swb + (mb + i) * 4);
  v8f asu = {};
#pragma unroll
  for (int q = 0; q < 2; ++q) {
    v16h af = afrag_lds(st, lane, q);    // cols q*32.. = heads 2q, 2q+1
    v8f c = {};
    c = wmma_f16(af, bfrag_g(wfrag, FB_SW2 + q, lane), c);
    int cc = n & 7;
    float bias = (n < 8) ? sb2[(2 * q) * 8 + cc] : sb2[(2 * q + 1) * 8 + cc];
#pragma unroll
    for (int i = 0; i < 8; ++i) {
      float wv = (n < 8) ? sw4[i][2 * q] : sw4[i][2 * q + 1];
      asu[i] += (c[i] + bias) * wv;
    }
  }
#pragma unroll
  for (int i = 0; i < 8; ++i) asu[i] += __shfl_xor(asu[i], 8, 32);

  // --- transpose updates into ob[ch][pixel] ---
#pragma unroll
  for (int i = 0; i < 8; ++i) ob[n * 16 + (mb + i)] = afu[i];
  if (n < 8) {
#pragma unroll
    for (int i = 0; i < 8; ++i) ob[(16 + n) * 16 + (mb + i)] = asu[i];
  }

  // --- final coalesced stores (lane = pixel) ---
  if (lane < 16) {
    int p = p0 + lane;
    float* outb = out + (size_t)b * CH * PIX;
#pragma unroll 4
    for (int ch = 0; ch < 16; ++ch)
      outb[ch * PIX + p] = (xb[ch * PIX + p] + ob[ch * 16 + lane] * scl) * aext;
#pragma unroll 4
    for (int ch = 16; ch < 24; ++ch)
      outb[ch * PIX + p] = (xb[ch * PIX + p] + ob[ch * 16 + lane]) * aext;
#pragma unroll
    for (int ch = 24; ch < 28; ++ch)
      outb[ch * PIX + p] = xb[ch * PIX + p];
  }
}

extern "C" void kernel_launch(void* const* d_in, const int* in_sizes, int n_in,
                              void* d_out, int out_size, void* d_ws, size_t ws_size,
                              hipStream_t stream) {
  (void)in_sizes; (void)n_in; (void)out_size; (void)ws_size;
  const float* x     = (const float*)d_in[0];
  const float* pw1   = (const float*)d_in[1];
  const float* pb1   = (const float*)d_in[2];
  const float* pw2   = (const float*)d_in[3];
  const float* pb2   = (const float*)d_in[4];
  const float* faw   = (const float*)d_in[5];
  const float* fab   = (const float*)d_in[6];
  const float* saw   = (const float*)d_in[7];
  const float* sab   = (const float*)d_in[8];
  const float* atemp = (const float*)d_in[9];
  const float* fW1   = (const float*)d_in[10];
  const float* fb1   = (const float*)d_in[11];
  const float* fW2   = (const float*)d_in[12];
  const float* fb2   = (const float*)d_in[13];
  const float* sW1   = (const float*)d_in[14];
  const float* sb1   = (const float*)d_in[15];
  const float* sW2   = (const float*)d_in[16];
  const float* sb2   = (const float*)d_in[17];
  const float* stw1  = (const float*)d_in[18];
  const float* stb1  = (const float*)d_in[19];
  const float* stw2  = (const float*)d_in[20];
  const float* stb2  = (const float*)d_in[21];
  const float* stw3  = (const float*)d_in[22];
  const float* stb3  = (const float*)d_in[23];
  float* out = (float*)d_out;
  float* hdr = (float*)d_ws;
  _Float16* wfrag = (_Float16*)((char*)d_ws + 256);

  k_prep<<<(NFRAG * 512 + 255) / 256, 256, 0, stream>>>(pw1, pw2, fW1, fW2,
                                                        sW1, sW2, wfrag);
  k_init<<<1, 32, 0, stream>>>(hdr);
  k_count<<<(NBAT * PIX) / 256, 256, 0, stream>>>(x, hdr);
  k_final<<<1, 32, 0, stream>>>(hdr);
  k_main<<<4096, 256, 0, stream>>>(x, pb1, pb2, faw, fab, saw, sab, atemp,
                                   fb1, fb2, sb1, sb2, stw1, stb1, stw2, stb2,
                                   stw3, stb3, wfrag, hdr, out);
}
